// RobustQuoteAttention_38362647888252
// MI455X (gfx1250) — matrix operand order repaired and checked by reading, not compiled, not run on hardware
//
#include <hip/hip_runtime.h>
#include <hip/hip_bf16.h>

// ---------------------------------------------------------------------------
// RobustQuoteAttention on MI455X (gfx1250, wave32, WMMA bf16 w/ f32 accum)
// B=16, N=256, C=768, G=4, H=12, d=64
// ---------------------------------------------------------------------------

typedef __attribute__((ext_vector_type(16))) __bf16 v16bf;
typedef __attribute__((ext_vector_type(8)))  float  v8f;

#define BB      16
#define NN      256
#define CC      768
#define GG      4
#define NHEADS  12
#define DD      64

// Optional CDNA5 async global->LDS path (builtin confirmed present; wants int*)
#if defined(__has_builtin)
#if __has_builtin(__builtin_amdgcn_global_load_async_to_lds_b32) && \
    __has_builtin(__builtin_amdgcn_s_wait_asynccnt)
#define HAVE_ASYNC_LDS 1
#endif
#endif

#if defined(HAVE_ASYNC_LDS)
typedef __attribute__((address_space(1))) int glb_int;
typedef __attribute__((address_space(3))) int lds_int;
__device__ __forceinline__ void async_g2l_b32(const void* g, void* l) {
    __builtin_amdgcn_global_load_async_to_lds_b32((glb_int*)g, (lds_int*)l, 0, 0);
}
#endif

// A/B fragment K order chosen here (contiguous, not the ISA table order; 16x32 A):
//   lanes 0-15 : VGPR0..7 hold K=0..15
//   lanes 16-31: VGPR0..7 hold K=16..31
__device__ __forceinline__ int a_kloc(int e, int grp) {
    return e + 16 * grp;
}

__device__ __forceinline__ v8f wmma_bf16(const v16bf& a, const v16bf& b, const v8f& c) {
    return __builtin_amdgcn_wmma_f32_16x16x32_bf16(false, a, false, b, (short)0, c,
                                                   false, false);
}

// ---------------------------------------------------------------------------
// K1: qv = anchors @ w_qv^T ; scatter into q16[G,H,N,D], v16[G,H,N,D] (bf16)
// ---------------------------------------------------------------------------
__global__ void __launch_bounds__(128)
qv_proj_kernel(const float* __restrict__ anchors,  // [G,N,C] row-major (G*N rows)
               const float* __restrict__ w_qv,     // [2C, C]
               __bf16* __restrict__ q16,
               __bf16* __restrict__ v16)
{
    __shared__ float At[16][33];
    __shared__ float Bt[64][33];
    const int tid  = threadIdx.x;
    const int lane = tid & 31;
    const int wave = tid >> 5;
    const int grp  = lane >> 4;
    const int lr   = lane & 15;

    const int m0 = blockIdx.x * 16;   // row in [0, G*N)
    const int nb = blockIdx.y * 64;   // col base in [0, 2C)

    v8f acc = {};
    for (int k0 = 0; k0 < CC; k0 += 32) {
        {   // stage A tile 16x32 (f32)
            const int row = tid >> 3, cg = (tid & 7) * 4;
            const float* src = anchors + (size_t)(m0 + row) * CC + k0 + cg;
            if (k0 + 32 < CC) __builtin_prefetch(src + 32, 0, 3);
            At[row][cg + 0] = src[0]; At[row][cg + 1] = src[1];
            At[row][cg + 2] = src[2]; At[row][cg + 3] = src[3];
        }
        {   // stage B tile 64x32: W[nb+n][k0+k]
            const int n = tid >> 1, hh = (tid & 1) * 16;
            const float* src = w_qv + (size_t)(nb + n) * CC + k0 + hh;
            #pragma unroll
            for (int j = 0; j < 16; ++j) Bt[n][hh + j] = src[j];
        }
        __syncthreads();
        v16bf a, b;
        #pragma unroll
        for (int e = 0; e < 16; ++e) {
            a[e] = (__bf16)At[lr][a_kloc(e, grp)];
            b[e] = (__bf16)Bt[wave * 16 + lr][grp * 16 + e];
        }
        acc = wmma_bf16(a, b, acc);
        __syncthreads();
    }
    // scatter: col j -> (two, h, dd); row m -> (g, n)
    const int j   = nb + wave * 16 + lr;
    const int two = j / CC;
    const int h   = (j % CC) / DD;
    const int dd  = j % DD;
    __bf16* dst = (two == 0) ? q16 : v16;
    #pragma unroll
    for (int r = 0; r < 8; ++r) {
        const int m = m0 + r + 8 * grp;
        const int g = m >> 8;
        const int n = m & 255;
        dst[(((size_t)(g * NHEADS + h)) * NN + n) * DD + dd] = (__bf16)acc[r];
    }
}

// ---------------------------------------------------------------------------
// K2: k = x @ w_k^T ; scatter into k16[B,H,N,D] (bf16)
// ---------------------------------------------------------------------------
__global__ void __launch_bounds__(128)
k_proj_kernel(const float* __restrict__ x,    // [B*N, C]
              const float* __restrict__ w_k,  // [C, C]
              __bf16* __restrict__ k16)
{
    __shared__ float At[16][33];
    __shared__ float Bt[64][33];
    const int tid  = threadIdx.x;
    const int lane = tid & 31;
    const int wave = tid >> 5;
    const int grp  = lane >> 4;
    const int lr   = lane & 15;

    const int m0 = blockIdx.x * 16;   // row in [0, B*N)
    const int nb = blockIdx.y * 64;   // col base in [0, C)

    v8f acc = {};
    for (int k0 = 0; k0 < CC; k0 += 32) {
        {
            const int row = tid >> 3, cg = (tid & 7) * 4;
            const float* src = x + (size_t)(m0 + row) * CC + k0 + cg;
            if (k0 + 32 < CC) __builtin_prefetch(src + 32, 0, 3);
            At[row][cg + 0] = src[0]; At[row][cg + 1] = src[1];
            At[row][cg + 2] = src[2]; At[row][cg + 3] = src[3];
        }
        {
            const int n = tid >> 1, hh = (tid & 1) * 16;
            const float* src = w_k + (size_t)(nb + n) * CC + k0 + hh;
            #pragma unroll
            for (int j = 0; j < 16; ++j) Bt[n][hh + j] = src[j];
        }
        __syncthreads();
        v16bf a, b;
        #pragma unroll
        for (int e = 0; e < 16; ++e) {
            a[e] = (__bf16)At[lr][a_kloc(e, grp)];
            b[e] = (__bf16)Bt[wave * 16 + lr][grp * 16 + e];
        }
        acc = wmma_bf16(a, b, acc);
        __syncthreads();
    }
    const int j  = nb + wave * 16 + lr;
    const int h  = j / DD;
    const int dd = j % DD;
    #pragma unroll
    for (int r = 0; r < 8; ++r) {
        const int m = m0 + r + 8 * grp;
        const int b = m >> 8;
        const int n = m & 255;
        k16[(((size_t)(b * NHEADS + h)) * NN + n) * DD + dd] = (__bf16)acc[r];
    }
}

// ---------------------------------------------------------------------------
// K3: scores S = q·k^T * scale for one (g,b,h, 16-row stripe); capture raw
//     S[0][0] for the class gate; row softmax; write f32 attn to d_out.
//     512 threads = 16 waves; wave w owns 16x16 tile at columns w*16.
// ---------------------------------------------------------------------------
__global__ void __launch_bounds__(512)
attn_softmax_kernel(const __bf16* __restrict__ q16,   // [G,H,N,D]
                    const __bf16* __restrict__ k16,   // [B,H,N,D]
                    float* __restrict__ attn_out,     // [G,B,H,N,N]
                    float* __restrict__ cls_raw)      // [G,B,H]
{
    __shared__ __bf16 qs[16][DD + 2];  // padded: 33-dword row stride, no bank conflict
    __shared__ float  S[16][NN + 1];   // padded: 257-dword row stride
    const int tid  = threadIdx.x;
    const int lane = tid & 31;
    const int wave = tid >> 5;
    const int grp  = lane >> 4;
    const int lr   = lane & 15;

    const int nblk = blockIdx.x;          // 0..15
    const int gbh  = blockIdx.y;          // 0..G*B*H-1
    const int g  = gbh / (BB * NHEADS);
    const int bh = gbh % (BB * NHEADS);
    const int b  = bh / NHEADS;
    const int h  = bh % NHEADS;
    const int n0 = nblk * 16;

    // stage q rows [n0, n0+16) x 64 into LDS: thread t copies dword t
    {
        const int row = tid >> 5, cu = tid & 31;  // 32 dwords per 64-elem row
        const uint32_t* src =
            (const uint32_t*)(q16 + ((size_t)(g * NHEADS + h) * NN + n0) * DD) +
            row * 32 + cu;
        uint32_t* dst = (uint32_t*)&qs[row][cu * 2];
#if defined(HAVE_ASYNC_LDS)
        async_g2l_b32(src, dst);
#else
        *dst = *src;
#endif
    }
#if defined(HAVE_ASYNC_LDS)
    __builtin_amdgcn_s_wait_asynccnt(0);
#endif
    __syncthreads();

    const __bf16* kb = k16 + (size_t)(b * NHEADS + h) * NN * DD;
    const int m0 = wave * 16;
    v8f acc = {};
    #pragma unroll
    for (int k0 = 0; k0 < DD; k0 += 32) {
        v16bf a, bfr;
        #pragma unroll
        for (int e = 0; e < 16; ++e) {
            a[e]   = qs[lr][k0 + a_kloc(e, grp)];
            bfr[e] = kb[(size_t)(m0 + lr) * DD + k0 + grp * 16 + e];  // k^T
        }
        acc = wmma_bf16(a, bfr, acc);
    }
    const float scale = 0.125f;   // 1/sqrt(64)
    #pragma unroll
    for (int r = 0; r < 8; ++r)
        S[r + 8 * grp][m0 + lr] = acc[r] * scale;
    __syncthreads();

    // class gate uses PRE-softmax (0,0) score
    if (nblk == 0 && tid == 0) cls_raw[gbh] = S[0][0];

    // row softmax: wave w reduces row w with wave32 shuffles
    {
        const int row = wave;
        float mx = -1e30f;
        #pragma unroll
        for (int c = 0; c < NN; c += 32) mx = fmaxf(mx, S[row][c + lane]);
        #pragma unroll
        for (int off = 16; off >= 1; off >>= 1) mx = fmaxf(mx, __shfl_xor(mx, off, 32));
        float ev[8];
        float sum = 0.f;
        #pragma unroll
        for (int i = 0; i < 8; ++i) {
            ev[i] = __expf(S[row][i * 32 + lane] - mx);
            sum += ev[i];
        }
        #pragma unroll
        for (int off = 16; off >= 1; off >>= 1) sum += __shfl_xor(sum, off, 32);
        const float inv = 1.f / sum;
        float* dst = attn_out +
            ((((size_t)g * BB + b) * NHEADS + h) * NN + (n0 + row)) * NN;
        #pragma unroll
        for (int i = 0; i < 8; ++i) dst[i * 32 + lane] = ev[i] * inv;
    }
}

// ---------------------------------------------------------------------------
// K4: attn_cls[b,h,:] = softmax_g(raw scores)
// ---------------------------------------------------------------------------
__global__ void cls_softmax_kernel(const float* __restrict__ cls_raw,  // [G,B,H]
                                   float* __restrict__ cls_out)        // [B,H,G]
{
    const int t = blockIdx.x * blockDim.x + threadIdx.x;
    if (t >= BB * NHEADS) return;
    const int b = t / NHEADS, h = t % NHEADS;
    float s[GG];
    float mx = -1e30f;
    #pragma unroll
    for (int g = 0; g < GG; ++g) {
        s[g] = cls_raw[(g * BB + b) * NHEADS + h];
        mx = fmaxf(mx, s[g]);
    }
    float sum = 0.f;
    #pragma unroll
    for (int g = 0; g < GG; ++g) { s[g] = __expf(s[g] - mx); sum += s[g]; }
    const float inv = 1.f / sum;
    #pragma unroll
    for (int g = 0; g < GG; ++g)
        cls_out[((size_t)b * NHEADS + h) * GG + g] = s[g] * inv;
}

// ---------------------------------------------------------------------------
// K5: y[b,n,h*64+dd] = sum_g cls[b,h,g] * (attn[g,b,h] @ v[g,h])[n,dd]
//     256 threads = 8 waves, one 16x16 tile per wave. attn f32 -> bf16 inline.
// ---------------------------------------------------------------------------
__global__ void __launch_bounds__(256)
av_combine_kernel(const float* __restrict__ attn,   // [G,B,H,N,N] post-softmax
                  const __bf16* __restrict__ v16,   // [G,H,N,D]
                  const float* __restrict__ cls,    // [B,H,G]
                  __bf16* __restrict__ y16)         // [B,N,C]
{
    const int tid  = threadIdx.x;
    const int lane = tid & 31;
    const int wave = tid >> 5;
    const int grp  = lane >> 4;
    const int lr   = lane & 15;

    const int t    = blockIdx.x * 8 + wave;    // tile id, 0..12287
    const int bh   = t >> 6;                   // 16 nblk * 4 dblk per (b,h)
    const int rem  = t & 63;
    const int nblk = rem >> 2;
    const int dblk = rem & 3;
    const int b = bh / NHEADS, h = bh % NHEADS;
    const int n0 = nblk * 16, d0 = dblk * 16;

    float accf[8] = {0.f, 0.f, 0.f, 0.f, 0.f, 0.f, 0.f, 0.f};
    for (int g = 0; g < GG; ++g) {
        const float cw = cls[((size_t)b * NHEADS + h) * GG + g];
        const float* arow = attn +
            ((((size_t)g * BB + b) * NHEADS + h) * NN + n0) * NN;
        const __bf16* vb = v16 + (size_t)(g * NHEADS + h) * NN * DD;
        v8f acc = {};
        for (int k0 = 0; k0 < NN; k0 += 32) {
            v16bf a, bfr;
            #pragma unroll
            for (int e = 0; e < 16; ++e) {
                a[e]   = (__bf16)arow[(size_t)lr * NN + k0 + a_kloc(e, grp)];
                bfr[e] = vb[(size_t)(k0 + grp * 16 + e) * DD + d0 + lr];
            }
            acc = wmma_bf16(a, bfr, acc);
        }
        #pragma unroll
        for (int r = 0; r < 8; ++r) accf[r] += cw * acc[r];
    }
    #pragma unroll
    for (int r = 0; r < 8; ++r) {
        const int n = n0 + r + 8 * grp;
        y16[((size_t)b * NN + n) * CC + h * DD + d0 + lr] = (__bf16)accf[r];
    }
}

// ---------------------------------------------------------------------------
// K6: out = y @ w_proj^T + b_proj  (f32 output)
// ---------------------------------------------------------------------------
__global__ void __launch_bounds__(128)
out_proj_kernel(const __bf16* __restrict__ y16,    // [B*N, C]
                const float* __restrict__ w_proj,  // [C, C]
                const float* __restrict__ b_proj,  // [C]
                float* __restrict__ out)           // [B*N, C]
{
    __shared__ __bf16 At[16][34];
    __shared__ float  Bt[64][33];
    const int tid  = threadIdx.x;
    const int lane = tid & 31;
    const int wave = tid >> 5;
    const int grp  = lane >> 4;
    const int lr   = lane & 15;

    const int m0 = blockIdx.x * 16;
    const int nb = blockIdx.y * 64;

    v8f acc = {};
    for (int k0 = 0; k0 < CC; k0 += 32) {
        {
            const int row = tid >> 3, cg = (tid & 7) * 4;
            const __bf16* src = y16 + (size_t)(m0 + row) * CC + k0 + cg;
            if (k0 + 32 < CC) __builtin_prefetch(src + 32, 0, 3);
            At[row][cg + 0] = src[0]; At[row][cg + 1] = src[1];
            At[row][cg + 2] = src[2]; At[row][cg + 3] = src[3];
        }
        {
            const int n = tid >> 1, hh = (tid & 1) * 16;
            const float* src = w_proj + (size_t)(nb + n) * CC + k0 + hh;
            #pragma unroll
            for (int j = 0; j < 16; ++j) Bt[n][hh + j] = src[j];
        }
        __syncthreads();
        v16bf a, bfr;
        #pragma unroll
        for (int e = 0; e < 16; ++e) {
            a[e]   = At[lr][a_kloc(e, grp)];
            bfr[e] = (__bf16)Bt[wave * 16 + lr][grp * 16 + e];
        }
        acc = wmma_bf16(a, bfr, acc);
        __syncthreads();
    }
    const int j = nb + wave * 16 + lr;
    const float bias = b_proj[j];
    #pragma unroll
    for (int r = 0; r < 8; ++r) {
        const int m = m0 + r + 8 * grp;
        out[(size_t)m * CC + j] = acc[r] + bias;
    }
}

// ---------------------------------------------------------------------------
extern "C" void kernel_launch(void* const* d_in, const int* in_sizes, int n_in,
                              void* d_out, int out_size, void* d_ws, size_t ws_size,
                              hipStream_t stream)
{
    (void)in_sizes; (void)n_in; (void)out_size; (void)ws_size;

    const float* x       = (const float*)d_in[0];  // [B,N,C]
    const float* anchors = (const float*)d_in[1];  // [G,N,C]
    const float* w_qv    = (const float*)d_in[2];  // [2C,C]
    const float* w_k     = (const float*)d_in[3];  // [C,C]
    const float* w_proj  = (const float*)d_in[4];  // [C,C]
    const float* b_proj  = (const float*)d_in[5];  // [C]

    // output layout: out [B,N,C] | attn_cls [B,H,G] | attn [G,B,H,N,N]
    float* out  = (float*)d_out;
    float* cls  = out + (size_t)BB * NN * CC;
    float* attn = cls + (size_t)BB * NHEADS * GG;

    // workspace: q16 | v16 | k16 | y16 | cls_raw  (~15.7 MB)
    __bf16* q16 = (__bf16*)d_ws;
    __bf16* v16 = q16 + (size_t)GG * NHEADS * NN * DD;
    __bf16* k16 = v16 + (size_t)GG * NHEADS * NN * DD;
    __bf16* y16 = k16 + (size_t)BB * NHEADS * NN * DD;
    float* cls_raw = (float*)(y16 + (size_t)BB * NN * CC);

    qv_proj_kernel<<<dim3(GG * NN / 16, 2 * CC / 64), 128, 0, stream>>>(
        anchors, w_qv, q16, v16);

    k_proj_kernel<<<dim3(BB * NN / 16, CC / 64), 128, 0, stream>>>(
        x, w_k, k16);

    attn_softmax_kernel<<<dim3(NN / 16, GG * BB * NHEADS), 512, 0, stream>>>(
        q16, k16, attn, cls_raw);

    cls_softmax_kernel<<<dim3(1), 192, 0, stream>>>(cls_raw, cls);

    av_combine_kernel<<<dim3((GG * BB * NHEADS * (NN / 16) * (DD / 16)) / 8), 256, 0,
                        stream>>>(attn, v16, cls, y16);

    out_proj_kernel<<<dim3(BB * NN / 16, CC / 64), 128, 0, stream>>>(
        y16, w_proj, b_proj, out);
}
